// VQVAETransformerFinetune_77077483094388
// MI455X (gfx1250) — compile-verified, matrix operand order checked by reading, not run on hardware
//
#include <hip/hip_runtime.h>
#include <hip/hip_bf16.h>

// ---------------------------------------------------------------------------
// VQVAE transformer decode for MI455X (gfx1250, wave32).
// GEMMs: bf16 WMMA (v_wmma_f32_16x16x32_bf16), 128x128x32 tiles, 8 waves/WG
// (4x2 wave grid, 32x64 per wave), double-buffered LDS with explicit
// register-level software pipelining: next tile is fetched into 4 distinct
// uint4 register quads BEFORE the WMMA block and stored to LDS AFTER it, so
// s_wait_loadcnt lands behind the matrix math instead of serializing HBM
// round-trips. Weights pre-transposed to [N,K] bf16 so A/B tiles stage
// identically and every fragment is two 16B ds_load_b128 chunks per lane.
// Activations consumed as GEMM-A are kept as bf16 mirrors (zbf/ob/h1).
// ---------------------------------------------------------------------------

typedef __bf16 v16bf  __attribute__((ext_vector_type(16)));
typedef __bf16 bf16x8 __attribute__((ext_vector_type(8)));
typedef float  v8f    __attribute__((ext_vector_type(8)));

#define BM 128
#define BN 128
#define BK 32
#define LDS_P 40   // LDS row stride (bf16): 80 B -> every 16B chunk aligned, bank-skewed

// ----------------------------- generic bf16 GEMM ---------------------------
// C[M,N] = A[M,K](bf16) @ Wt[N,K](bf16, pre-transposed) + bias[N]
// Cf (f32 out) and/or Cb (bf16 out), optional relu. M%128==N%128==K%32==0.
__global__ __launch_bounds__(256) void gemm_bf16_k(
    const __bf16* __restrict__ A, const __bf16* __restrict__ Wt,
    const float* __restrict__ bias, float* __restrict__ Cf,
    __bf16* __restrict__ Cb, int M, int N, int K, int relu)
{
  __shared__ __align__(16) __bf16 sA[2][BM][LDS_P];
  __shared__ __align__(16) __bf16 sB[2][BN][LDS_P];   // transposed: [n][k]

  const int tid  = threadIdx.x;
  const int wid  = tid >> 5;
  const int lane = tid & 31;
  const int lh   = lane >> 4;   // lane half
  const int l16  = lane & 15;
  const int bm   = blockIdx.y * BM;
  const int bn   = blockIdx.x * BN;
  const int wm   = (wid >> 1) * 32;  // 4 waves along M
  const int wn   = (wid & 1) * 64;   // 2 waves along N

  v8f acc[2][4];
  #pragma unroll
  for (int i = 0; i < 2; ++i)
    #pragma unroll
    for (int j = 0; j < 4; ++j)
      #pragma unroll
      for (int r = 0; r < 8; ++r) acc[i][j][r] = 0.f;

  // Per-thread staging coords: tile = 128 rows x 32 k bf16, 16 elems/thread.
  const int sr0 = (tid * 8) >> 5;          // row, first half (0..63)
  const int sc0 = (tid * 8) & 31;          // k offset, multiple of 8
  const int sr1 = ((tid + 256) * 8) >> 5;  // row, second half (64..127)
  const int sc1 = ((tid + 256) * 8) & 31;

  // Distinct register quads -> all 4 global loads issue before any wait.
  uint4 ra0, ra1, rb0, rb1;
  auto fetch = [&](int k0) {
    ra0 = *(const uint4*)(A  + (size_t)(bm + sr0) * K + k0 + sc0);
    ra1 = *(const uint4*)(A  + (size_t)(bm + sr1) * K + k0 + sc1);
    rb0 = *(const uint4*)(Wt + (size_t)(bn + sr0) * K + k0 + sc0);
    rb1 = *(const uint4*)(Wt + (size_t)(bn + sr1) * K + k0 + sc1);
  };
  auto stage = [&](int bi) {
    *(uint4*)(&sA[bi][sr0][sc0]) = ra0;
    *(uint4*)(&sA[bi][sr1][sc1]) = ra1;
    *(uint4*)(&sB[bi][sr0][sc0]) = rb0;
    *(uint4*)(&sB[bi][sr1][sc1]) = rb1;
  };

  fetch(0);
  stage(0);
  __syncthreads();

  const int nk = K / BK;
  for (int kt = 0; kt < nk; ++kt) {
    const int cur = kt & 1;
    if (kt + 1 < nk) fetch((kt + 1) * BK);   // issue loads, no wait yet

    // Fragment = two contiguous 16B chunks: K[lh*8..+7] and K[16+lh*8..+7].
    v16bf af[2], bfr[4];
    #pragma unroll
    for (int mt = 0; mt < 2; ++mt) {
      const int mr = wm + mt * 16 + l16;
      const bf16x8 lo = *(const bf16x8*)(&sA[cur][mr][lh * 8]);
      const bf16x8 hi = *(const bf16x8*)(&sA[cur][mr][16 + lh * 8]);
      af[mt] = __builtin_shufflevector(lo, hi, 0,1,2,3,4,5,6,7,8,9,10,11,12,13,14,15);
    }
    #pragma unroll
    for (int nt = 0; nt < 4; ++nt) {
      const int nc = wn + nt * 16 + l16;
      const bf16x8 lo = *(const bf16x8*)(&sB[cur][nc][lh * 8]);
      const bf16x8 hi = *(const bf16x8*)(&sB[cur][nc][16 + lh * 8]);
      bfr[nt] = __builtin_shufflevector(lo, hi, 0,1,2,3,4,5,6,7,8,9,10,11,12,13,14,15);
    }
    #pragma unroll
    for (int mt = 0; mt < 2; ++mt)
      #pragma unroll
      for (int nt = 0; nt < 4; ++nt)
        acc[mt][nt] = __builtin_amdgcn_wmma_f32_16x16x32_bf16(
            false, af[mt], false, bfr[nt], (short)0, acc[mt][nt], false, false);

    if (kt + 1 < nk) stage(cur ^ 1);         // wait lands here, after WMMAs
    __syncthreads();
  }

  // C/D layout: element r -> m = tile_m + r + 8*lh, n = tile_n + (lane&15)
  #pragma unroll
  for (int mt = 0; mt < 2; ++mt) {
    #pragma unroll
    for (int nt = 0; nt < 4; ++nt) {
      const int gm0 = bm + wm + mt * 16 + lh * 8;
      const int gn  = bn + wn + nt * 16 + l16;
      const float bv = bias[gn];
      #pragma unroll
      for (int r = 0; r < 8; ++r) {
        float x = acc[mt][nt][r] + bv;
        if (relu) x = fmaxf(x, 0.f);
        if (Cf) Cf[(size_t)(gm0 + r) * N + gn] = x;
        if (Cb) Cb[(size_t)(gm0 + r) * N + gn] = (__bf16)x;
      }
    }
  }
}

// -------------------- weight convert + transpose (f32->bf16) ---------------
// src: L x [K,N] f32; dst: L x [N,K] bf16. K,N multiples of 32.
__global__ __launch_bounds__(256) void cvt_transpose_k(
    const float* __restrict__ src, __bf16* __restrict__ dst, int K, int N)
{
  __shared__ float tile[32][33];
  const int l  = blockIdx.z;
  const int n0 = blockIdx.x * 32;
  const int k0 = blockIdx.y * 32;
  const int tx = threadIdx.x & 31;
  const int ty = threadIdx.x >> 5;          // 0..7
  const float* s = src + (size_t)l * K * N;
  __bf16* d = dst + (size_t)l * N * K;
  #pragma unroll
  for (int i = 0; i < 32; i += 8)
    tile[ty + i][tx] = s[(size_t)(k0 + ty + i) * N + n0 + tx];
  __syncthreads();
  #pragma unroll
  for (int i = 0; i < 32; i += 8)
    d[(size_t)(n0 + ty + i) * K + k0 + tx] = (__bf16)tile[tx][ty + i];
}

// ------------------------------- attention ---------------------------------
// qkv: [Nn,288,1536] f32 = [q|k|v]; H=8, dh=64. One wave = 32 queries of one
// (n,h); K/V addresses are wave-uniform -> scalar loads; per-lane softmax.
// Output written as bf16 (only consumed as GEMM-A by the Wo projection).
__global__ __launch_bounds__(32) void attn_k(const float* __restrict__ qkv,
                                             __bf16* __restrict__ o, int cur_len)
{
  const int qb = blockIdx.x % 9;
  const int h  = (blockIdx.x / 9) & 7;
  const int n  = blockIdx.x / 72;
  const int q  = qb * 32 + threadIdx.x;

  const size_t base = (size_t)n * 288 * 1536;
  const float* qp = qkv + base + (size_t)q * 1536 + h * 64;

  float qv[64];
  #pragma unroll
  for (int d = 0; d < 64; ++d) qv[d] = qp[d] * 0.125f;  // 1/sqrt(64)

  float mx = -3.0e38f, ssum = 0.f;
  float acc[64];
  #pragma unroll
  for (int d = 0; d < 64; ++d) acc[d] = 0.f;

  const int je = qb * 32 + 32;
  const int jend = (je < cur_len) ? je : cur_len;   // uniform trip count
  for (int j = 0; j < jend; ++j) {
    const float* kp = qkv + base + (size_t)j * 1536 + 512 + h * 64;
    const float* vp = kp + 512;
    float s = 0.f;
    #pragma unroll
    for (int d = 0; d < 64; ++d) s = fmaf(qv[d], kp[d], s);
    if (j > q) s = -1.0e30f;                          // causal mask (per-lane)
    const float mn   = fmaxf(mx, s);
    const float corr = __expf(mx - mn);
    const float p    = __expf(s - mn);
    ssum = ssum * corr + p;
    #pragma unroll
    for (int d = 0; d < 64; ++d) acc[d] = fmaf(acc[d], corr, p * vp[d]);
    mx = mn;
  }
  const float inv = 1.f / ssum;
  __bf16* op = o + ((size_t)n * 288 + q) * 512 + h * 64;
  #pragma unroll
  for (int d8 = 0; d8 < 8; ++d8) {
    bf16x8 pk;
    #pragma unroll
    for (int j = 0; j < 8; ++j) pk[j] = (__bf16)(acc[d8 * 8 + j] * inv);
    *(bf16x8*)(op + d8 * 8) = pk;
  }
}

// --------------------------- residual + layernorm --------------------------
// z = LN(z + t); also refresh the bf16 mirror zb.
__global__ __launch_bounds__(256) void ln_res_k(float* __restrict__ z,
    __bf16* __restrict__ zb, const float* __restrict__ t,
    const float* __restrict__ s, const float* __restrict__ b)
{
  __shared__ float red[256];
  const int tid = threadIdx.x;
  float* zr = z + (size_t)blockIdx.x * 512;
  __bf16* zbr = zb + (size_t)blockIdx.x * 512;
  const float* tr = t + (size_t)blockIdx.x * 512;
  const float u0 = zr[tid] + tr[tid];
  const float u1 = zr[tid + 256] + tr[tid + 256];
  red[tid] = u0 + u1;
  __syncthreads();
  #pragma unroll
  for (int off = 128; off > 0; off >>= 1) {
    if (tid < off) red[tid] += red[tid + off];
    __syncthreads();
  }
  const float mean = red[0] * (1.f / 512.f);
  __syncthreads();
  const float d0 = u0 - mean, d1 = u1 - mean;
  red[tid] = d0 * d0 + d1 * d1;
  __syncthreads();
  #pragma unroll
  for (int off = 128; off > 0; off >>= 1) {
    if (tid < off) red[tid] += red[tid + off];
    __syncthreads();
  }
  const float inv = rsqrtf(red[0] * (1.f / 512.f) + 1e-5f);
  const float r0 = d0 * inv * s[tid] + b[tid];
  const float r1 = d1 * inv * s[tid + 256] + b[tid + 256];
  zr[tid] = r0;
  zr[tid + 256] = r1;
  zbr[tid] = (__bf16)r0;
  zbr[tid + 256] = (__bf16)r1;
}

// ------------------------------ small kernels ------------------------------
// argmax over K=512 of codebook_probs[B=8,K=512,Tc=256,C=7] -> idx0[Nn=56,Tc]
__global__ void init_argmax_k(const float* __restrict__ probs, int* __restrict__ idx0)
{
  const int gid = blockIdx.x * 256 + threadIdx.x;
  if (gid >= 56 * 256) return;
  const int t = gid & 255;
  const int n = gid >> 8;        // n = b*7 + c
  const int b = n / 7, c = n % 7;
  const float* p = probs + (size_t)b * 512 * 256 * 7 + (size_t)t * 7 + c;
  float best = -3.4e38f; int bi = 0;
  for (int k = 0; k < 512; ++k) {
    const float v = p[(size_t)k * 256 * 7];
    if (v > best) { best = v; bi = k; }   // first-max tie break (jnp.argmax)
  }
  idx0[n * 256 + t] = bi;
}

// buf[n,t] = codebook[idx0]@W_proj + b_proj + W_pos[t] for t<256; zero beyond.
__global__ __launch_bounds__(128) void init_embed_k(const int* __restrict__ idx0,
    const float* __restrict__ codebook, const float* __restrict__ W_proj,
    const float* __restrict__ b_proj, const float* __restrict__ W_pos,
    float* __restrict__ buf, __bf16* __restrict__ zb)
{
  const int bid = blockIdx.x;          // Nn*288
  const int t = bid % 288;
  const int n = bid / 288;
  float* out = buf + (size_t)bid * 512;
  __bf16* outb = zb + (size_t)bid * 512;
  if (t >= 256) {
    for (int d = threadIdx.x; d < 512; d += 128) { out[d] = 0.f; outb[d] = (__bf16)0.f; }
    return;
  }
  const int idx = idx0[n * 256 + t];
  const float* e = codebook + (size_t)idx * 64;
  for (int d = threadIdx.x; d < 512; d += 128) {
    float v = b_proj[d] + W_pos[(size_t)t * 512 + d];
    #pragma unroll
    for (int i = 0; i < 64; ++i) v = fmaf(e[i], W_proj[(size_t)i * 512 + d], v);
    out[d] = v;
    outb[d] = (__bf16)v;
  }
}

// logits = z[cur_len-1] @ W_head + b_head; argmax -> gen[step*56 + n]
// (softmax is monotone; gen is only consumed via argmax in the reference)
__global__ __launch_bounds__(256) void head_argmax_k(const float* __restrict__ buf,
    const float* __restrict__ W_head, const float* __restrict__ b_head,
    int* __restrict__ gen, int step, int cur_len)
{
  __shared__ float sz[512];
  __shared__ float sval[256];
  __shared__ int   sidx[256];
  const int n = blockIdx.x, t = threadIdx.x;
  const float* zr = buf + ((size_t)n * 288 + (cur_len - 1)) * 512;
  sz[t] = zr[t];
  sz[t + 256] = zr[t + 256];
  __syncthreads();
  float a0 = b_head[t], a1 = b_head[t + 256];
  for (int d = 0; d < 512; ++d) {
    const float zd = sz[d];
    a0 = fmaf(zd, W_head[(size_t)d * 512 + t], a0);
    a1 = fmaf(zd, W_head[(size_t)d * 512 + t + 256], a1);
  }
  float bv; int bi;
  if (a0 >= a1) { bv = a0; bi = t; } else { bv = a1; bi = t + 256; }
  sval[t] = bv; sidx[t] = bi;
  __syncthreads();
  #pragma unroll
  for (int off = 128; off > 0; off >>= 1) {
    if (t < off) {
      const float v2 = sval[t + off]; const int i2 = sidx[t + off];
      if (v2 > sval[t] || (v2 == sval[t] && i2 < sidx[t])) { sval[t] = v2; sidx[t] = i2; }
    }
    __syncthreads();
  }
  if (t == 0) gen[step * 56 + n] = sidx[0];
}

// buf[n, cur_len] = codebook[nidx[n]] @ W_proj + b_proj + W_pos[cur_len]
__global__ __launch_bounds__(128) void embed_next_k(const int* __restrict__ nidx,
    const float* __restrict__ codebook, const float* __restrict__ W_proj,
    const float* __restrict__ b_proj, const float* __restrict__ W_pos,
    float* __restrict__ buf, __bf16* __restrict__ zb, int cur_len)
{
  const int n = blockIdx.x;
  const int idx = nidx[n];
  const float* e = codebook + (size_t)idx * 64;
  float* out = buf + ((size_t)n * 288 + cur_len) * 512;
  __bf16* outb = zb + ((size_t)n * 288 + cur_len) * 512;
  for (int d = threadIdx.x; d < 512; d += 128) {
    float v = b_proj[d] + W_pos[(size_t)cur_len * 512 + d];
    #pragma unroll
    for (int i = 0; i < 64; ++i) v = fmaf(e[i], W_proj[(size_t)i * 512 + d], v);
    out[d] = v;
    outb[d] = (__bf16)v;
  }
}

// dec: q = codebook[gen]; out[b, s*4+k, c] = q @ W_dec[:,k] + b_dec
__global__ void decode_k(const int* __restrict__ gen, const float* __restrict__ codebook,
                         const float* __restrict__ W_dec, const float* __restrict__ b_dec,
                         float* __restrict__ out)
{
  const int gid = blockIdx.x * 256 + threadIdx.x;
  if (gid >= 8 * 7 * 32) return;
  const int s = gid & 31;
  const int c = (gid >> 5) % 7;
  const int b = gid / (32 * 7);
  const int idx = gen[s * 56 + b * 7 + c];
  const float* q = codebook + (size_t)idx * 64;
  const float bd = b_dec[0];
  #pragma unroll
  for (int k = 0; k < 4; ++k) {
    float v = bd;
    #pragma unroll
    for (int e = 0; e < 64; ++e) v = fmaf(q[e], W_dec[e * 4 + k], v);
    out[((size_t)b * 128 + s * 4 + k) * 7 + c] = v;
  }
}

// ------------------------------- host driver -------------------------------
extern "C" void kernel_launch(void* const* d_in, const int* in_sizes, int n_in,
                              void* d_out, int out_size, void* d_ws, size_t ws_size,
                              hipStream_t stream)
{
  (void)in_sizes; (void)n_in; (void)out_size; (void)ws_size;
  const int Tc = 256, total = 288, Nn = 56, D = 512, H3 = 1536, F = 2048;
  const int steps = 32, L = 4;
  const int Mrows = Nn * total;  // 16128 (divisible by 128)

  const float* codebook_probs = (const float*)d_in[0];
  const float* codebook = (const float*)d_in[1];
  const float* W_proj   = (const float*)d_in[2];
  const float* b_proj   = (const float*)d_in[3];
  const float* W_pos    = (const float*)d_in[4];
  const float* Wqkv     = (const float*)d_in[5];
  const float* bqkv     = (const float*)d_in[6];
  const float* Wo       = (const float*)d_in[7];
  const float* bo       = (const float*)d_in[8];
  const float* ln1_s    = (const float*)d_in[9];
  const float* ln1_b    = (const float*)d_in[10];
  const float* W1       = (const float*)d_in[11];
  const float* b1       = (const float*)d_in[12];
  const float* W2       = (const float*)d_in[13];
  const float* b2       = (const float*)d_in[14];
  const float* ln2_s    = (const float*)d_in[15];
  const float* ln2_b    = (const float*)d_in[16];
  const float* W_head   = (const float*)d_in[17];
  const float* b_head   = (const float*)d_in[18];
  const float* W_dec    = (const float*)d_in[19];
  const float* b_dec    = (const float*)d_in[20];
  float* out = (float*)d_out;

  // Workspace carve-up (~290 MB total)
  char* ws = (char*)d_ws;
  size_t off = 0;
  auto take = [&](size_t bytes) -> char* {
    char* p = ws + off;
    off += (bytes + 255) & ~(size_t)255;
    return p;
  };
  float*  buf  = (float*)take((size_t)Mrows * D * 4);     // z / zp master (f32)
  __bf16* zbf  = (__bf16*)take((size_t)Mrows * D * 2);    // bf16 mirror of z
  float*  qkv  = (float*)take((size_t)Mrows * H3 * 4);    // f32, read by attention
  __bf16* ob   = (__bf16*)take((size_t)Mrows * D * 2);    // attention out (GEMM-A only)
  float*  tmp  = (float*)take((size_t)Mrows * D * 4);     // pre-LN GEMM output
  __bf16* h1   = (__bf16*)take((size_t)Mrows * F * 2);    // FFN hidden (GEMM-A only)
  __bf16* Wqkv_b = (__bf16*)take((size_t)L * D * H3 * 2); // all weights as [N,K] bf16
  __bf16* Wo_b   = (__bf16*)take((size_t)L * D * D * 2);
  __bf16* W1_b   = (__bf16*)take((size_t)L * D * F * 2);
  __bf16* W2_b   = (__bf16*)take((size_t)L * F * D * 2);
  int* idx0 = (int*)take((size_t)Nn * Tc * 4);
  int* gen  = (int*)take((size_t)steps * Nn * 4);

  // Weight convert+transpose (every call: deterministic, no caching)
  cvt_transpose_k<<<dim3(H3 / 32, D / 32, L), 256, 0, stream>>>(Wqkv, Wqkv_b, D, H3);
  cvt_transpose_k<<<dim3(D / 32, D / 32, L), 256, 0, stream>>>(Wo, Wo_b, D, D);
  cvt_transpose_k<<<dim3(F / 32, D / 32, L), 256, 0, stream>>>(W1, W1_b, D, F);
  cvt_transpose_k<<<dim3(D / 32, F / 32, L), 256, 0, stream>>>(W2, W2_b, F, D);

  // Initial buffer: argmax -> embed, zeros beyond Tc
  init_argmax_k<<<(Nn * Tc + 255) / 256, 256, 0, stream>>>(codebook_probs, idx0);
  init_embed_k<<<Mrows, 128, 0, stream>>>(idx0, codebook, W_proj, b_proj, W_pos, buf, zbf);

  auto gemm = [&](const __bf16* A, const __bf16* W, const float* bias,
                  float* Cf, __bf16* Cb, int M, int N, int K, int relu) {
    dim3 g(N / BN, M / BM);
    gemm_bf16_k<<<g, 256, 0, stream>>>(A, W, bias, Cf, Cb, M, N, K, relu);
  };

  for (int s = 0; s < steps; ++s) {
    const int cur_len = Tc + s;
    for (int l = 0; l < L; ++l) {
      gemm(zbf, Wqkv_b + (size_t)l * D * H3, bqkv + (size_t)l * H3,
           qkv, nullptr, Mrows, H3, D, 0);
      attn_k<<<Nn * 8 * (total / 32), 32, 0, stream>>>(qkv, ob, cur_len);
      gemm(ob, Wo_b + (size_t)l * D * D, bo + (size_t)l * D,
           tmp, nullptr, Mrows, D, D, 0);
      ln_res_k<<<Mrows, 256, 0, stream>>>(buf, zbf, tmp,
                                          ln1_s + (size_t)l * D, ln1_b + (size_t)l * D);
      gemm(zbf, W1_b + (size_t)l * D * F, b1 + (size_t)l * F,
           nullptr, h1, Mrows, F, D, 1);
      gemm(h1, W2_b + (size_t)l * F * D, b2 + (size_t)l * D,
           tmp, nullptr, Mrows, D, F, 0);
      ln_res_k<<<Mrows, 256, 0, stream>>>(buf, zbf, tmp,
                                          ln2_s + (size_t)l * D, ln2_b + (size_t)l * D);
    }
    head_argmax_k<<<Nn, 256, 0, stream>>>(buf, W_head, b_head, gen, s, cur_len);
    embed_next_k<<<Nn, 128, 0, stream>>>(gen + (size_t)s * Nn, codebook, W_proj,
                                         b_proj, W_pos, buf, zbf, cur_len);
  }

  decode_k<<<7, 256, 0, stream>>>(gen, codebook, W_dec, b_dec, out);
}
